// ReviewAutoencoder_74723841016403
// MI455X (gfx1250) — matrix operand order repaired and checked by reading, not compiled, not run on hardware
//
#include <hip/hip_runtime.h>
#include <hip/hip_bf16.h>

// ---------------------------------------------------------------------------
// ReviewAutoencoder on MI455X (gfx1250): WMMA f16 GEMMs + register-resident
// LSTM recurrence.  H=128, E=128, V=20000, B=16, S=256.
// ---------------------------------------------------------------------------

typedef _Float16 h16;
typedef __attribute__((ext_vector_type(16))) _Float16 v16h;
typedef __attribute__((ext_vector_type(8)))  _Float16 v8h;
typedef __attribute__((ext_vector_type(8)))  float    v8f;

#define HD   128          // hidden dim
#define G4   512          // 4*H
#define BB   16           // batch
#define SS   256          // seq len
#define VV   20000        // vocab
#define MTOT (BB * SS)    // 4096 rows in all big GEMMs

// ---- WMMA fragment helpers (layouts per CDNA5 ISA §7.12.2) ------------------

// A fragment (16x32 f16, MxK): lane l -> row M=l%16; halves:
//   [0..7]  = K = kbase + (l/16)*8 + {0..7}
//   [8..15] = K = kbase + 16 + (l/16)*8 + {0..7}
__device__ __forceinline__ v16h frag_a(const h16* __restrict__ A, int lda,
                                       int kbase, int lane) {
  const int m = lane & 15;
  const int g = (lane >> 4) & 1;
  const h16* p = A + m * lda + kbase + g * 8;
  v8h lo = *(const v8h*)(p);
  v8h hi = *(const v8h*)(p + 16);
  v16h a;
#pragma unroll
  for (int i = 0; i < 8; ++i) { a[i] = lo[i]; a[i + 8] = hi[i]; }
  return a;
}

// B fragment (32x16 f16, KxN) where B[k][n] = W[n][k], W row-major (N,K):
// lane l -> column N = n0 + l%16; [0..7] K = kbase + (l/16)*8 + {0..7},
// [8..15] K = kbase + 16 + (l/16)*8 + {0..7}.
__device__ __forceinline__ v16h frag_b(const h16* __restrict__ W, int ldw,
                                       int n0, int kbase, int lane) {
  const int n = n0 + (lane & 15);
  const int g = (lane >> 4) & 1;
  const h16* p = W + (long)n * ldw + kbase + g * 8;
  v8h lo = *(const v8h*)(p);
  v8h hi = *(const v8h*)(p + 16);
  v16h b;
#pragma unroll
  for (int i = 0; i < 8; ++i) { b[i] = lo[i]; b[i + 8] = hi[i]; }
  return b;
}

__device__ __forceinline__ v8f wmma32(v16h a, v16h b, v8f c) {
  return __builtin_amdgcn_wmma_f32_16x16x32_f16(false, a, false, b,
                                                (short)0, c, false, false);
}

__device__ __forceinline__ float sigf(float x) {
  return 1.0f / (1.0f + __expf(-x));
}

// ---------------------------------------------------------------------------
// Generic GEMM: out[M,N](f32) = A16[M,128] * W16[N,128]^T + bias1 + bias2
// One 16x16 output tile per wave, K=128 -> 4 x v_wmma_f32_16x16x32_f16.
// N is compile-time so the 8 row-strided stores fold into immediate offsets.
// NT=true uses non-temporal stores (logits are write-once, keep L2 clean).
// ---------------------------------------------------------------------------
template <int N, bool NT>
__global__ void __launch_bounds__(256) wmma_gemm_k128(
    const h16* __restrict__ A, const h16* __restrict__ W,
    const float* __restrict__ bias1, const float* __restrict__ bias2,
    float* __restrict__ out, int M) {
  const int lane = threadIdx.x & 31;
  const int wave = threadIdx.x >> 5;
  constexpr int ntiles = N >> 4;
  const long total = (long)(M >> 4) * ntiles;
  const long tile = (long)blockIdx.x * 8 + wave;
  if (tile >= total) return;                 // wave-uniform: EXEC stays all-1s
  const int mt = (int)(tile / ntiles);
  const int nt = (int)(tile % ntiles);
  const int m0 = mt << 4, n0 = nt << 4;

  const h16* Atile = A + (long)m0 * HD;
  v8f acc = {};
#pragma unroll
  for (int kk = 0; kk < 4; ++kk) {
    v16h a = frag_a(Atile, HD, kk * 32, lane);
    v16h b = frag_b(W, HD, n0, kk * 32, lane);
    acc = wmma32(a, b, acc);
  }
  const int mrow = (lane >> 4) * 8;
  const int nc = n0 + (lane & 15);
  float bsum = 0.0f;
  if (bias1) bsum += bias1[nc];
  if (bias2) bsum += bias2[nc];
  float* p = out + (long)(m0 + mrow) * N + nc;   // single base address
#pragma unroll
  for (int v = 0; v < 8; ++v) {
    float val = acc[v] + bsum;
    if (NT)
      __builtin_nontemporal_store(val, p + (long)v * N);  // imm offsets
    else
      p[(long)v * N] = val;
  }
}

// ---------------------------------------------------------------------------
// Sequential LSTM layer, one workgroup (8 waves), batch=16 == one WMMA M-tile.
// Wave w owns hidden slice [16w, 16w+16): its i/f/g/o tiles sit at gate
// offsets {0,128,256,384}+16w.  Cell state c lives in the WMMA C layout in
// registers for all T steps.  W_hh is staged once as f16 in LDS (128 KB) so
// per-step B fragments are ds_loads; xp gate inputs are register
// double-buffered (step t prefetches t+1) to hide global-load latency under
// the WMMA + transcendental work.
//   xp : (16, T, 512) f32 pre-projected gates incl. both biases
//   Whh: (512, 128) f32 row-major
// ---------------------------------------------------------------------------
__global__ void __launch_bounds__(256) lstm_seq(
    const float* __restrict__ xp, const float* __restrict__ Whh,
    const float* __restrict__ h0,       // (16,128) f32 or nullptr (zeros)
    h16* __restrict__ hs_out,           // (16,T,128) f16 or nullptr
    float* __restrict__ hT_out,         // (16,128) f32 or nullptr
    int T) {
  __shared__ __align__(32) h16 Wlds[G4 * HD];   // 128 KB: W_hh as f16
  __shared__ __align__(32) h16 hbuf[BB * HD];   // 4 KB: current h
  const int tid = threadIdx.x;
  const int lane = tid & 31;
  const int wave = tid >> 5;

  // cooperative f32->f16 conversion of W_hh into LDS (256 elems/thread)
  for (int i = tid; i < G4 * HD; i += 256) Wlds[i] = (h16)Whh[i];
  // init h in LDS (2048 halves / 256 threads = 8 each)
#pragma unroll
  for (int i = 0; i < 8; ++i) {
    int idx = tid * 8 + i;
    hbuf[idx] = h0 ? (h16)h0[idx] : (h16)0.0f;
  }
  __syncthreads();

  const int j = wave * 16;            // hidden-slice base
  const int mrow = (lane >> 4) * 8;   // C-layout row offset (0 or 8)
  const int nl = lane & 15;           // C-layout column
  float c[8], hreg[8];
#pragma unroll
  for (int v = 0; v < 8; ++v) { c[v] = 0.0f; hreg[v] = 0.0f; }

  v8f xpc[4], xpn[4];
  auto load_xp = [&](int t, v8f* dst) {
#pragma unroll
    for (int q = 0; q < 4; ++q) {
      const int n0 = q * HD + j;
#pragma unroll
      for (int v = 0; v < 8; ++v)
        dst[q][v] = xp[(long)(mrow + v) * (T * G4) + (long)t * G4 + n0 + nl];
    }
  };
  load_xp(0, xpc);

  for (int t = 0; t < T; ++t) {
    // prefetch next step's gate inputs (consumed only at loop bottom)
    const int tn = (t + 1 < T) ? t + 1 : t;
    load_xp(tn, xpn);

    // h fragments (shared by all 4 gate tiles of this wave)
    v16h a0 = frag_a(hbuf, HD, 0, lane);
    v16h a1 = frag_a(hbuf, HD, 32, lane);
    v16h a2 = frag_a(hbuf, HD, 64, lane);
    v16h a3 = frag_a(hbuf, HD, 96, lane);

    v8f gate[4];
#pragma unroll
    for (int q = 0; q < 4; ++q) {
      const int n0 = q * HD + j;
      v8f acc = xpc[q];
      acc = wmma32(a0, frag_b(Wlds, HD, n0, 0, lane), acc);
      acc = wmma32(a1, frag_b(Wlds, HD, n0, 32, lane), acc);
      acc = wmma32(a2, frag_b(Wlds, HD, n0, 64, lane), acc);
      acc = wmma32(a3, frag_b(Wlds, HD, n0, 96, lane), acc);
      gate[q] = acc;
    }
    __syncthreads();   // everyone done reading old h
#pragma unroll
    for (int v = 0; v < 8; ++v) {
      float ig = sigf(gate[0][v]);
      float fg = sigf(gate[1][v]);
      float gg = tanhf(gate[2][v]);
      float og = sigf(gate[3][v]);
      c[v] = fg * c[v] + ig * gg;
      float h = og * tanhf(c[v]);
      hreg[v] = h;
      hbuf[(mrow + v) * HD + j + nl] = (h16)h;
      if (hs_out)
        hs_out[(long)(mrow + v) * (T * HD) + (long)t * HD + j + nl] = (h16)h;
    }
    __syncthreads();   // new h visible before next step's reads
#pragma unroll
    for (int q = 0; q < 4; ++q) xpc[q] = xpn[q];
  }
  if (hT_out) {
#pragma unroll
    for (int v = 0; v < 8; ++v)
      hT_out[(mrow + v) * HD + j + nl] = hreg[v];
  }
}

// ---------------------------------------------------------------------------
// Small helper kernels
// ---------------------------------------------------------------------------
__global__ void cvt_f16(const float* __restrict__ in, h16* __restrict__ out,
                        int n) {
  int i = blockIdx.x * 256 + threadIdx.x;
  if (i < n) out[i] = (h16)in[i];
}

// emb16[b*S+t][e] = (f16) embedding[token_ids[b*S+t]][e]
__global__ void embed_gather(const int* __restrict__ ids,
                             const float* __restrict__ emb,
                             h16* __restrict__ out) {
  int i = blockIdx.x * 256 + threadIdx.x;
  if (i >= MTOT * HD) return;
  int bs = i >> 7, e = i & 127;
  out[i] = (h16)emb[(long)ids[bs] * HD + e];
}

// dxp0[bs][g] = dec_W_ih0[g][clean_id] + b_ih0[g] + b_hh0[g]  (one-hot GEMM)
__global__ void dec_onehot_gather(const int* __restrict__ ids,
                                  const float* __restrict__ Wih0,
                                  const float* __restrict__ b1,
                                  const float* __restrict__ b2,
                                  float* __restrict__ out) {
  long i = (long)blockIdx.x * 256 + threadIdx.x;
  if (i >= (long)MTOT * G4) return;
  int bs = (int)(i >> 9), g = (int)(i & 511);
  int id = ids[bs];
  if (id <= 0) id = 0;
  out[i] = Wih0[(long)g * VV + id] + b1[g] + b2[g];
}

// ---------------------------------------------------------------------------
extern "C" void kernel_launch(void* const* d_in, const int* in_sizes, int n_in,
                              void* d_out, int out_size, void* d_ws,
                              size_t ws_size, hipStream_t stream) {
  (void)in_sizes; (void)n_in; (void)out_size; (void)ws_size;
  const int*   tok       = (const int*)d_in[0];
  const float* embedding = (const float*)d_in[1];
  const float* eWih0 = (const float*)d_in[2];
  const float* eWhh0 = (const float*)d_in[3];
  const float* eBih0 = (const float*)d_in[4];
  const float* eBhh0 = (const float*)d_in[5];
  const float* eWih1 = (const float*)d_in[6];
  const float* eWhh1 = (const float*)d_in[7];
  const float* eBih1 = (const float*)d_in[8];
  const float* eBhh1 = (const float*)d_in[9];
  const float* dWih0 = (const float*)d_in[10];
  const float* dWhh0 = (const float*)d_in[11];
  const float* dBih0 = (const float*)d_in[12];
  const float* dBhh0 = (const float*)d_in[13];
  const float* dWih1 = (const float*)d_in[14];
  const float* dWhh1 = (const float*)d_in[15];
  const float* dBih1 = (const float*)d_in[16];
  const float* dBhh1 = (const float*)d_in[17];
  const float* fcW   = (const float*)d_in[18];
  const float* fcB   = (const float*)d_in[19];
  float* logits = (float*)d_out;

  // ---- workspace carve-up (256B aligned) ----
  char* base = (char*)d_ws;
  size_t off = 0;
  auto carve = [&](size_t bytes) -> void* {
    void* p = base + off;
    off = (off + bytes + 255) & ~(size_t)255;
    return p;
  };
  h16*   emb16  = (h16*)carve((size_t)MTOT * HD * 2);    // 1 MB
  h16*   hs0    = (h16*)carve((size_t)MTOT * HD * 2);    // 1 MB
  h16*   dhs0   = (h16*)carve((size_t)MTOT * HD * 2);    // 1 MB
  h16*   dhs1   = (h16*)carve((size_t)MTOT * HD * 2);    // 1 MB
  float* xpbuf  = (float*)carve((size_t)MTOT * G4 * 4);  // 8 MB (reused 4x)
  float* latent = (float*)carve((size_t)BB * HD * 4);
  h16* eWih0_16 = (h16*)carve((size_t)G4 * HD * 2);
  h16* eWih1_16 = (h16*)carve((size_t)G4 * HD * 2);
  h16* dWih1_16 = (h16*)carve((size_t)G4 * HD * 2);
  h16* fcW_16   = (h16*)carve((size_t)VV * HD * 2);      // 5 MB

  const int NW = G4 * HD;  // 65536 elements per small weight
  // ---- weight conversions (f32 -> f16); W_hh converted inside lstm_seq ----
  cvt_f16<<<(NW + 255) / 256, 256, 0, stream>>>(eWih0, eWih0_16, NW);
  cvt_f16<<<(NW + 255) / 256, 256, 0, stream>>>(eWih1, eWih1_16, NW);
  cvt_f16<<<(NW + 255) / 256, 256, 0, stream>>>(dWih1, dWih1_16, NW);
  cvt_f16<<<(VV * HD + 255) / 256, 256, 0, stream>>>(fcW, fcW_16, VV * HD);

  const int xpBlocks = (int)(((long)(MTOT / 16) * (G4 / 16) + 7) / 8);  // 1024

  // ---- encoder ----
  embed_gather<<<(MTOT * HD + 255) / 256, 256, 0, stream>>>(tok, embedding,
                                                            emb16);
  // xp0 = emb @ W_ih0^T + b_ih0 + b_hh0   : 4096x512x128
  wmma_gemm_k128<G4, false><<<xpBlocks, 256, 0, stream>>>(
      emb16, eWih0_16, eBih0, eBhh0, xpbuf, MTOT);
  lstm_seq<<<1, 256, 0, stream>>>(xpbuf, eWhh0, nullptr, hs0, nullptr, SS);
  // xp1 = hs0 @ W_ih1^T + biases
  wmma_gemm_k128<G4, false><<<xpBlocks, 256, 0, stream>>>(
      hs0, eWih1_16, eBih1, eBhh1, xpbuf, MTOT);
  lstm_seq<<<1, 256, 0, stream>>>(xpbuf, eWhh1, nullptr, nullptr, latent, SS);

  // ---- decoder ----
  // dxp0 = one-hot GEMM == column gather of dec_W_ih0
  dec_onehot_gather<<<(int)(((long)MTOT * G4 + 255) / 256), 256, 0, stream>>>(
      tok, dWih0, dBih0, dBhh0, xpbuf);
  lstm_seq<<<1, 256, 0, stream>>>(xpbuf, dWhh0, latent, dhs0, nullptr, SS);
  // dxp1 = dhs0 @ dec_W_ih1^T + biases
  wmma_gemm_k128<G4, false><<<xpBlocks, 256, 0, stream>>>(
      dhs0, dWih1_16, dBih1, dBhh1, xpbuf, MTOT);
  lstm_seq<<<1, 256, 0, stream>>>(xpbuf, dWhh1, latent, dhs1, nullptr, SS);

  // ---- logits = dhs1 @ fc_W^T + fc_b : 4096 x 20000 x 128 (HBM-store bound)
  {
    long tiles = (long)(MTOT / 16) * (VV / 16);  // 256 * 1250 = 320000
    wmma_gemm_k128<VV, true><<<(int)((tiles + 7) / 8), 256, 0, stream>>>(
        dhs1, fcW_16, fcB, nullptr, logits, MTOT);
  }
}
